// HierarchicalGNN_31172872634961
// MI455X (gfx1250) — matrix operand order repaired and checked by reading, not compiled, not run on hardware
//
#include <hip/hip_runtime.h>
#include <math.h>

typedef __bf16 bf16_t;
typedef __attribute__((ext_vector_type(16))) __bf16 v16bf;
typedef __attribute__((ext_vector_type(8)))  __bf16 v8bf;
typedef __attribute__((ext_vector_type(8)))  float  v8f;

#define NN 6000
#define MMOT 16
#define EMOT 32
#define NE 96000
#define MD 256
#define CD 512
#define CEILDIV(a,b) (((a)+(b)-1)/(b))

// ---------------------------------------------------------------------------
// WMMA helpers: bf16 A (16xK row-major), bf16 B from pre-transposed weights
// (Kout x KinPad row-major => contiguous per-lane loads), f32 C/D.
// A layout (16-bit, 16x32): lane row = lane&15, half = lane>>4;
//   elem i -> K = kk + i + half*16  (contiguous 16, matches B)
// B layout (32x16): lane col = lane&15; elem i -> K = kk + i + half*16 (contiguous 16)
// D layout: vgpr r -> M = r + half*8, N = lane&15
// ---------------------------------------------------------------------------
__device__ __forceinline__ v16bf wmma_load_a(const bf16_t* A, int ldk, int kk, int lane) {
  int row = lane & 15;
  int half = lane >> 4;
  const bf16_t* p = A + (size_t)row * ldk + kk + half * 16;
  v8bf lo = *(const v8bf*)p;
  v8bf hi = *(const v8bf*)(p + 8);
  return __builtin_shufflevector(lo, hi, 0,1,2,3,4,5,6,7,8,9,10,11,12,13,14,15);
}

__device__ __forceinline__ v16bf wmma_load_b(const bf16_t* Wt, int n0, int kinpad, int kk, int lane) {
  int col = n0 + (lane & 15);
  int half = lane >> 4;
  return *(const v16bf*)(Wt + (size_t)col * kinpad + kk + half * 16);
}

__device__ __forceinline__ v8f wmma_bf16(v16bf a, v16bf b, v8f c) {
  return __builtin_amdgcn_wmma_f32_16x16x32_bf16(false, a, false, b, (short)0, c, false, false);
}

__device__ __forceinline__ void atomicMaxF(float* addr, float val) {
  unsigned int* ia = (unsigned int*)addr;
  unsigned int old = *ia;
  while (__uint_as_float(old) < val) {
    unsigned int assumed = old;
    old = atomicCAS(ia, assumed, __float_as_uint(val));
    if (old == assumed) break;
  }
}

// ---------------------------------------------------------------------------
// Generic bf16 WMMA GEMM: C(rows x kout) = A(rows x kin) @ W + bias
// One wave computes a 16x64 output strip (4 WMMA tiles sharing one A load):
// 2.5 b128 loads per WMMA instead of 4, and 4x fewer A re-reads across kout.
// flags: 1=relu, 2=accumulate into C. kout must be a multiple of 64.
// ---------------------------------------------------------------------------
__device__ __forceinline__ void gemm_store_tile(v8f acc, float* __restrict__ C,
                                                bf16_t* __restrict__ Cbf,
                                                const float* __restrict__ bias,
                                                int m0, int n0, int kout, int flags, int lane) {
  int n = n0 + (lane & 15);
  int half = lane >> 4;
  float bb = bias ? bias[n] : 0.f;
#pragma unroll
  for (int r = 0; r < 8; ++r) {
    int m = m0 + r + half * 8;
    size_t idx = (size_t)m * kout + n;
    float v = acc[r] + bb;
    if (flags & 2) v += C[idx];
    if (flags & 1) v = fmaxf(v, 0.f);
    C[idx] = v;
    if (Cbf) Cbf[idx] = (bf16_t)v;
  }
}

__global__ void __launch_bounds__(32)
k_gemm(const bf16_t* __restrict__ A, const bf16_t* __restrict__ Wt,
       const float* __restrict__ bias, float* __restrict__ C,
       bf16_t* __restrict__ Cbf, int kin, int kout, int flags) {
  int lane = threadIdx.x;
  int m0 = blockIdx.x * 16;
  int n0 = blockIdx.y * 64;
  v8f acc0 = {}, acc1 = {}, acc2 = {}, acc3 = {};
  const bf16_t* Arow = A + (size_t)m0 * kin;
  for (int kk = 0; kk < kin; kk += 32) {
    v16bf a  = wmma_load_a(Arow, kin, kk, lane);
    v16bf b0 = wmma_load_b(Wt, n0,      kin, kk, lane);
    v16bf b1 = wmma_load_b(Wt, n0 + 16, kin, kk, lane);
    v16bf b2 = wmma_load_b(Wt, n0 + 32, kin, kk, lane);
    v16bf b3 = wmma_load_b(Wt, n0 + 48, kin, kk, lane);
    acc0 = wmma_bf16(a, b0, acc0);
    acc1 = wmma_bf16(a, b1, acc1);
    acc2 = wmma_bf16(a, b2, acc2);
    acc3 = wmma_bf16(a, b3, acc3);
  }
  gemm_store_tile(acc0, C, Cbf, bias, m0, n0,      kout, flags, lane);
  gemm_store_tile(acc1, C, Cbf, bias, m0, n0 + 16, kout, flags, lane);
  gemm_store_tile(acc2, C, Cbf, bias, m0, n0 + 32, kout, flags, lane);
  gemm_store_tile(acc3, C, Cbf, bias, m0, n0 + 48, kout, flags, lane);
}

// ---------------------------------------------------------------------------
// Weight convert+transpose: src f32 (kin x kout) -> dst bf16 (kout x kinpad)
// ---------------------------------------------------------------------------
__global__ void k_wt(const float* __restrict__ src, bf16_t* __restrict__ dst,
                     int kin, int kout, int kinpad) {
  int t = blockIdx.x * blockDim.x + threadIdx.x;
  int total = kout * kinpad;
  if (t >= total) return;
  int co = t / kinpad, ki = t - co * kinpad;
  float v = (ki < kin) ? src[(size_t)ki * kout + co] : 0.f;
  dst[t] = (bf16_t)v;
}

__global__ void k_fillf(float* p, float v, int n) {
  int t = blockIdx.x * blockDim.x + threadIdx.x;
  if (t < n) p[t] = v;
}

__global__ void k_addvec(const float* a, const float* b, float* o, int n) {
  int t = blockIdx.x * blockDim.x + threadIdx.x;
  if (t < n) o[t] = a[t] + b[t];
}

// ---------------------------------------------------------------------------
// Per-motif kernel: GraphConv (norm both, edge-weighted) + GATv2, one wave/motif.
// Produces motif_feats (f32 + bf16 copies).
// ---------------------------------------------------------------------------
__global__ void __launch_bounds__(32)
k_motif(const float* __restrict__ atom_x, const float* __restrict__ ew_all,
        const int* __restrict__ src_all, const int* __restrict__ dst_all,
        const bf16_t* __restrict__ Wac_t, const float* __restrict__ b_ac,
        const bf16_t* __restrict__ Wl_t, const bf16_t* __restrict__ Wr_t,
        const float* __restrict__ attn_a, const float* __restrict__ b_gat,
        float* __restrict__ motif_feats, bf16_t* __restrict__ motif_feats_bf) {
  int g = blockIdx.x;
  int lane = threadIdx.x;

  __shared__ float  s_tmp[MMOT * 13];
  __shared__ float  s_acc[MMOT * 13];
  __shared__ float  s_degi[MMOT], s_dego[MMOT];
  __shared__ __align__(32) bf16_t s_hbf[MMOT * 32];
  __shared__ __align__(32) bf16_t s_ahbf[MMOT * MD];
  __shared__ float  s_el[MMOT * MD];
  __shared__ float  s_er[MMOT * MD];
  __shared__ int    s_src[EMOT], s_dst[EMOT];
  __shared__ float  s_ew[EMOT];
  __shared__ float  s_sc[EMOT * 4];
  __shared__ float  s_al[EMOT * 4];
  __shared__ float  s_m[MMOT * 4], s_z[MMOT * 4];

  const float* ax = atom_x + (size_t)g * MMOT * 13;
  s_src[lane] = src_all[(size_t)g * EMOT + lane];
  s_dst[lane] = dst_all[(size_t)g * EMOT + lane];
  s_ew[lane]  = ew_all[(size_t)g * EMOT + lane];
  if (lane < MMOT) { s_degi[lane] = 0.f; s_dego[lane] = 0.f; }
  for (int i = lane; i < MMOT * 13; i += 32) s_acc[i] = 0.f;
  __syncthreads();

  atomicAdd(&s_dego[s_src[lane]], 1.f);
  atomicAdd(&s_degi[s_dst[lane]], 1.f);
  __syncthreads();

  for (int i = lane; i < MMOT * 13; i += 32) {
    int r = i / 13;
    s_tmp[i] = ax[i] * rsqrtf(fmaxf(s_dego[r], 1.f));
  }
  __syncthreads();

  { // edge-weighted message passing
    int sR = s_src[lane], dR = s_dst[lane];
    float w = s_ew[lane];
    for (int c = 0; c < 13; ++c)
      atomicAdd(&s_acc[dR * 13 + c], s_tmp[sR * 13 + c] * w);
  }
  __syncthreads();

  for (int i = lane; i < MMOT * 32; i += 32) { // h -> bf16, pad K 13->32
    int r = i >> 5, c = i & 31;
    float hv = (c < 13) ? s_acc[r * 13 + c] * rsqrtf(fmaxf(s_degi[r], 1.f)) : 0.f;
    s_hbf[i] = (bf16_t)hv;
  }
  __syncthreads();

  int nloc = lane & 15, half = lane >> 4;

  { // ah = relu(h @ W_ac + b_ac): 16 tiles, single K=32 step each
    v16bf a = wmma_load_a(s_hbf, 32, 0, lane);
    for (int t = 0; t < 16; ++t) {
      int n0 = t * 16;
      v16bf b = wmma_load_b(Wac_t, n0, 32, 0, lane);
      v8f c = {};
      c = wmma_bf16(a, b, c);
      int col = n0 + nloc;
      float bb = b_ac[col];
#pragma unroll
      for (int r = 0; r < 8; ++r)
        s_ahbf[(r + half * 8) * MD + col] = (bf16_t)fmaxf(c[r] + bb, 0.f);
    }
  }
  __syncthreads();

  // el = ah @ Wl, er = ah @ Wr (16x256 each)
  for (int t = 0; t < 16; ++t) {
    int n0 = t * 16;
    v8f cl = {}, cr = {};
    for (int kk = 0; kk < MD; kk += 32) {
      v16bf a  = wmma_load_a(s_ahbf, MD, kk, lane);
      v16bf bl = wmma_load_b(Wl_t, n0, MD, kk, lane);
      cl = wmma_bf16(a, bl, cl);
      v16bf br = wmma_load_b(Wr_t, n0, MD, kk, lane);
      cr = wmma_bf16(a, br, cr);
    }
    int col = n0 + nloc;
#pragma unroll
    for (int r = 0; r < 8; ++r) {
      s_el[(r + half * 8) * MD + col] = cl[r];
      s_er[(r + half * 8) * MD + col] = cr[r];
    }
  }
  __syncthreads();

  { // GATv2 scores: lane = edge
    int sR = s_src[lane], dR = s_dst[lane];
    for (int h = 0; h < 4; ++h) {
      float s = 0.f;
      for (int dd = 0; dd < 64; ++dd) {
        int idx = h * 64 + dd;
        float v = s_el[sR * MD + idx] + s_er[dR * MD + idx];
        v = v > 0.f ? v : 0.2f * v; // leaky_relu 0.2
        s += v * attn_a[idx];
      }
      s_sc[lane * 4 + h] = s;
    }
  }
  __syncthreads();

  // segment softmax over dst (16 nodes x 4 heads = 64 cells)
  for (int it = 0; it < 2; ++it) {
    int cell = lane + it * 32;
    int node = cell & 15, hh = cell >> 4;
    float mv = -3.4e38f;
    for (int e = 0; e < EMOT; ++e)
      if (s_dst[e] == node) mv = fmaxf(mv, s_sc[e * 4 + hh]);
    if (mv < -1e37f) mv = 0.f;
    float zv = 0.f;
    for (int e = 0; e < EMOT; ++e)
      if (s_dst[e] == node) zv += __expf(s_sc[e * 4 + hh] - mv);
    s_m[cell] = mv;
    s_z[cell] = zv;
  }
  __syncthreads();

  {
    int dR = s_dst[lane];
    for (int h = 0; h < 4; ++h) {
      int cell = h * 16 + dR;
      s_al[lane * 4 + h] = __expf(s_sc[lane * 4 + h] - s_m[cell]) / fmaxf(s_z[cell], 1e-16f);
    }
  }
  __syncthreads();

  // mean over nodes of scatter(al*el[src]) == (1/16) * sum over edges
  for (int j = 0; j < 8; ++j) {
    int dd = j * 32 + lane;
    int h = dd >> 6;
    float acc = 0.f;
    for (int e = 0; e < EMOT; ++e)
      acc += s_al[e * 4 + h] * s_el[s_src[e] * MD + dd];
    float f = acc * (1.f / 16.f) + b_gat[dd];
    motif_feats[(size_t)g * MD + dd] = f;
    motif_feats_bf[(size_t)g * MD + dd] = (bf16_t)f;
  }
}

// ---------------------------------------------------------------------------
// Chain-level elementwise / scatter kernels
// ---------------------------------------------------------------------------
__global__ void k_gin_agg(const float* __restrict__ mf, const int* __restrict__ src,
                          const int* __restrict__ dst, float* __restrict__ agg, int E) {
  int e = blockIdx.x;
  if (e >= E) return;
  int d = threadIdx.x;
  atomicAdd(&agg[(size_t)dst[e] * MD + d], mf[(size_t)src[e] * MD + d]);
}

__global__ void k_add_conv(const float* a, const float* b, bf16_t* obf, int n) {
  int t = blockIdx.x * blockDim.x + threadIdx.x;
  if (t < n) obf[t] = (bf16_t)(a[t] + b[t]);
}

__global__ void k_edge_max(const float* __restrict__ tth, const float* __restrict__ pph,
                           const int* __restrict__ src, const int* __restrict__ dst,
                           float* __restrict__ h_edge, int E) {
  int e = blockIdx.x;
  if (e >= E) return;
  int s = src[e], d0 = dst[e];
  for (int it = 0; it < 2; ++it) {
    int d = threadIdx.x + it * 256;
    float msg = tth[(size_t)s * CD + d] - tth[(size_t)d0 * CD + d] + pph[(size_t)d0 * CD + d];
    atomicMaxF(&h_edge[(size_t)d0 * CD + d], msg);
  }
}

__global__ void k_edge_fix(float* p, int n) {
  int t = blockIdx.x * blockDim.x + threadIdx.x;
  if (t < n) { float v = p[t]; if (!isfinite(v)) p[t] = 0.f; }
}

// TransformerConv score: one wave per edge, 4 heads x 128 dims
__global__ void k_tr_score(const float* __restrict__ q, const float* __restrict__ k,
                           const float* __restrict__ eattr, const float* __restrict__ We,
                           const float* __restrict__ be, const int* __restrict__ src,
                           const int* __restrict__ dst, float* __restrict__ sc, int E) {
  int wid = threadIdx.x >> 5, lane = threadIdx.x & 31;
  int e = blockIdx.x * 8 + wid;
  if (e >= E) return;
  int s = src[e], d0 = dst[e];
  float e0 = eattr[e * 5 + 0], e1 = eattr[e * 5 + 1], e2 = eattr[e * 5 + 2];
  float e3 = eattr[e * 5 + 3], e4 = eattr[e * 5 + 4];
  float partial = 0.f;
  int dbase = lane * 16;
#pragma unroll
  for (int i = 0; i < 16; ++i) {
    int dd = dbase + i;
    float ea = be[dd] + e0 * We[dd] + e1 * We[512 + dd] + e2 * We[1024 + dd]
             + e3 * We[1536 + dd] + e4 * We[2048 + dd];
    partial += q[(size_t)d0 * CD + dd] * (k[(size_t)s * CD + dd] + ea);
  }
  partial += __shfl_xor(partial, 1, 32);
  partial += __shfl_xor(partial, 2, 32);
  partial += __shfl_xor(partial, 4, 32);
  if ((lane & 7) == 0)
    sc[e * 4 + (lane >> 3)] = partial * 0.08838834764831845f; // 1/sqrt(128)
}

__global__ void k_tr_max(const float* __restrict__ sc, const int* __restrict__ dst,
                         float* __restrict__ m, int E) {
  int t = blockIdx.x * blockDim.x + threadIdx.x;
  if (t >= E * 4) return;
  int e = t >> 2, h = t & 3;
  atomicMaxF(&m[dst[e] * 4 + h], sc[t]);
}

__global__ void k_tr_exp(float* __restrict__ sc, const int* __restrict__ dst,
                         const float* __restrict__ m, float* __restrict__ z, int E) {
  int t = blockIdx.x * blockDim.x + threadIdx.x;
  if (t >= E * 4) return;
  int e = t >> 2, h = t & 3;
  float mm = m[dst[e] * 4 + h];
  if (!isfinite(mm)) mm = 0.f;
  float ex = __expf(sc[t] - mm);
  sc[t] = ex;
  atomicAdd(&z[dst[e] * 4 + h], ex);
}

__global__ void k_tr_agg(const float* __restrict__ v, const float* __restrict__ eattr,
                         const float* __restrict__ We, const float* __restrict__ be,
                         const int* __restrict__ src, const int* __restrict__ dst,
                         const float* __restrict__ sc, const float* __restrict__ z,
                         float* __restrict__ h_tr, int E) {
  int wid = threadIdx.x >> 5, lane = threadIdx.x & 31;
  int e = blockIdx.x * 8 + wid;
  if (e >= E) return;
  int s = src[e], d0 = dst[e];
  int h = lane >> 3;
  float a = sc[e * 4 + h] / fmaxf(z[d0 * 4 + h], 1e-16f);
  float e0 = eattr[e * 5 + 0], e1 = eattr[e * 5 + 1], e2 = eattr[e * 5 + 2];
  float e3 = eattr[e * 5 + 3], e4 = eattr[e * 5 + 4];
  int dbase = lane * 16;
#pragma unroll
  for (int i = 0; i < 16; ++i) {
    int dd = dbase + i;
    float ea = be[dd] + e0 * We[dd] + e1 * We[512 + dd] + e2 * We[1024 + dd]
             + e3 * We[1536 + dd] + e4 * We[2048 + dd];
    atomicAdd(&h_tr[(size_t)d0 * CD + dd], a * (v[(size_t)s * CD + dd] + ea));
  }
}

__global__ void k_chinit(const float* h_tr, const float* vnode, float* ch, bf16_t* chbf, int n) {
  int t = blockIdx.x * blockDim.x + threadIdx.x;
  if (t < n) {
    float v = h_tr[t] + vnode[t & (CD - 1)];
    ch[t] = v;
    chbf[t] = (bf16_t)v;
  }
}

// gated 4-way fusion: one wave per node
__global__ void k_gate(const float* __restrict__ hg, const float* __restrict__ he,
                       const float* __restrict__ ht, const float* __restrict__ ch,
                       const float* __restrict__ Wg, const float* __restrict__ bg,
                       float* __restrict__ out, int N) {
  int wid = threadIdx.x >> 5, lane = threadIdx.x & 31;
  int n = blockIdx.x * 8 + wid;
  if (n >= N) return;
  const float* rows[4] = { hg + (size_t)n * CD, he + (size_t)n * CD,
                           ht + (size_t)n * CD, ch + (size_t)n * CD };
  float l0 = 0.f, l1 = 0.f, l2 = 0.f, l3 = 0.f;
  for (int j = lane; j < 4 * CD; j += 32) {
    float xv = rows[j >> 9][j & (CD - 1)];
    const float* w = Wg + j * 4;
    l0 += xv * w[0]; l1 += xv * w[1]; l2 += xv * w[2]; l3 += xv * w[3];
  }
  for (int off = 16; off; off >>= 1) {
    l0 += __shfl_xor(l0, off, 32);
    l1 += __shfl_xor(l1, off, 32);
    l2 += __shfl_xor(l2, off, 32);
    l3 += __shfl_xor(l3, off, 32);
  }
  l0 += bg[0]; l1 += bg[1]; l2 += bg[2]; l3 += bg[3];
  float mx = fmaxf(fmaxf(l0, l1), fmaxf(l2, l3));
  float x0 = __expf(l0 - mx), x1 = __expf(l1 - mx), x2 = __expf(l2 - mx), x3 = __expf(l3 - mx);
  float zs = x0 + x1 + x2 + x3;
  float g0 = x0 / zs, g1 = x1 / zs, g2 = x2 / zs, g3 = x3 / zs;
  for (int dd = lane; dd < CD; dd += 32)
    out[(size_t)n * CD + dd] = g0 * rows[0][dd] + g1 * rows[1][dd] + g2 * rows[2][dd] + g3 * rows[3][dd];
}

__global__ void k_reduce(const float* __restrict__ out, float* __restrict__ rsum,
                         float* __restrict__ rmax, int n) {
  int t = blockIdx.x * blockDim.x + threadIdx.x;
  if (t >= n) return;
  int d = t & (CD - 1);
  float v = out[t];
  atomicAdd(&rsum[d], v);
  atomicMaxF(&rmax[d], v);
}

// readout + property head (single block of 1024 threads)
__global__ void __launch_bounds__(1024)
k_head(const float* __restrict__ rsum, const float* __restrict__ rmax,
       const float* __restrict__ W1, const float* __restrict__ b1,
       const float* __restrict__ bn_g, const float* __restrict__ bn_b,
       const float* __restrict__ W2, const float* __restrict__ b2,
       float* __restrict__ prop, float invN) {
  __shared__ float emb[1024];
  __shared__ float h1[1024];
  int t = threadIdx.x;
  emb[t] = (t < 512) ? rsum[t] * invN : rmax[t - 512];
  __syncthreads();
  float acc = b1[t];
  for (int j = 0; j < 1024; ++j) acc += emb[j] * W1[(size_t)j * 1024 + t];
  acc = acc * (bn_g[t] * 0.99999500003749969f) + bn_b[t]; // 1/sqrt(1+1e-5)
  h1[t] = fmaxf(acc, 0.f);
  __syncthreads();
  if (t < 29) {
    float p = b2[t];
    for (int j = 0; j < 1024; ++j) p += h1[j] * W2[(size_t)j * 29 + t];
    prop[t] = p;
  }
}

// ---------------------------------------------------------------------------
extern "C" void kernel_launch(void* const* d_in, const int* in_sizes, int n_in,
                              void* d_out, int out_size, void* d_ws, size_t ws_size,
                              hipStream_t stream) {
  const float* atom_x      = (const float*)d_in[0];
  const float* motif_ew    = (const float*)d_in[1];
  const float* chain_eattr = (const float*)d_in[2];
  const float* W_ac = (const float*)d_in[3];
  const float* b_ac = (const float*)d_in[4];
  const float* Wl   = (const float*)d_in[5];
  const float* Wr   = (const float*)d_in[6];
  const float* attn_a = (const float*)d_in[7];
  const float* b_gat  = (const float*)d_in[8];
  const float* W_gin  = (const float*)d_in[9];
  const float* b_gin  = (const float*)d_in[10];
  const float* W_th = (const float*)d_in[11];
  const float* b_th = (const float*)d_in[12];
  const float* W_ph = (const float*)d_in[13];
  const float* b_ph = (const float*)d_in[14];
  const float* Wq = (const float*)d_in[15];
  const float* bq = (const float*)d_in[16];
  const float* Wk = (const float*)d_in[17];
  const float* bk = (const float*)d_in[18];
  const float* Wv = (const float*)d_in[19];
  const float* bvv = (const float*)d_in[20];
  const float* We = (const float*)d_in[21];
  const float* be = (const float*)d_in[22];
  const float* Wskip = (const float*)d_in[23];
  const float* bskip = (const float*)d_in[24];
  const float* vnode = (const float*)d_in[25];
  const float* Wv_c = (const float*)d_in[26];
  const float* bv_c = (const float*)d_in[27];
  const float* Wo_c = (const float*)d_in[28];
  const float* bo_c = (const float*)d_in[29];
  const float* W_gate = (const float*)d_in[30];
  const float* b_gate = (const float*)d_in[31];
  const float* W1 = (const float*)d_in[32];
  const float* b1 = (const float*)d_in[33];
  const float* bn_g = (const float*)d_in[34];
  const float* bn_b = (const float*)d_in[35];
  const float* W2 = (const float*)d_in[36];
  const float* b2 = (const float*)d_in[37];
  const int* motif_src = (const int*)d_in[38];
  const int* motif_dst = (const int*)d_in[39];
  const int* chain_src = (const int*)d_in[40];
  const int* chain_dst = (const int*)d_in[41];

  char* wp = (char*)d_ws;
  auto alloc = [&](size_t bytes) -> void* {
    void* p = (void*)wp;
    wp += (bytes + 255) & ~(size_t)255;
    return p;
  };

  // bf16 transposed weights
  bf16_t* Wac_t   = (bf16_t*)alloc((size_t)256 * 32 * 2);
  bf16_t* Wl_t    = (bf16_t*)alloc((size_t)256 * 256 * 2);
  bf16_t* Wr_t    = (bf16_t*)alloc((size_t)256 * 256 * 2);
  bf16_t* Wgin_t  = (bf16_t*)alloc((size_t)512 * 256 * 2);
  bf16_t* Wth_t   = (bf16_t*)alloc((size_t)512 * 512 * 2);
  bf16_t* Wph_t   = (bf16_t*)alloc((size_t)512 * 512 * 2);
  bf16_t* Wq_t    = (bf16_t*)alloc((size_t)512 * 512 * 2);
  bf16_t* Wk_t    = (bf16_t*)alloc((size_t)512 * 512 * 2);
  bf16_t* Wv_t    = (bf16_t*)alloc((size_t)512 * 512 * 2);
  bf16_t* Wskip_t = (bf16_t*)alloc((size_t)512 * 512 * 2);
  bf16_t* Wvc_t   = (bf16_t*)alloc((size_t)3 * 512 * 512 * 2);
  bf16_t* Woc_t   = (bf16_t*)alloc((size_t)3 * 512 * 512 * 2);
  // activations
  float*  mf      = (float*)alloc((size_t)NN * MD * 4);
  bf16_t* mf_bf   = (bf16_t*)alloc((size_t)NN * MD * 2);
  float*  agg     = (float*)alloc((size_t)NN * MD * 4);
  bf16_t* s_bf    = (bf16_t*)alloc((size_t)NN * MD * 2);
  float*  hgin    = (float*)alloc((size_t)NN * CD * 4);
  bf16_t* hgin_bf = (bf16_t*)alloc((size_t)NN * CD * 2);
  float*  b_comb  = (float*)alloc((size_t)CD * 4);
  float*  t_th    = (float*)alloc((size_t)NN * CD * 4);
  float*  p_ph    = (float*)alloc((size_t)NN * CD * 4);
  float*  h_edge  = (float*)alloc((size_t)NN * CD * 4);
  float*  qb      = (float*)alloc((size_t)NN * CD * 4);
  float*  kb      = (float*)alloc((size_t)NN * CD * 4);
  float*  vb      = (float*)alloc((size_t)NN * CD * 4);
  float*  sc      = (float*)alloc((size_t)NE * 4 * 4);
  float*  mseg    = (float*)alloc((size_t)NN * 4 * 4);
  float*  zseg    = (float*)alloc((size_t)NN * 4 * 4);
  float*  h_tr    = (float*)alloc((size_t)NN * CD * 4);
  float*  ch      = (float*)alloc((size_t)NN * CD * 4);
  bf16_t* ch_bf   = (bf16_t*)alloc((size_t)NN * CD * 2);
  float*  tmp     = (float*)alloc((size_t)NN * CD * 4);
  bf16_t* tmp_bf  = (bf16_t*)alloc((size_t)NN * CD * 2);
  float*  rsum    = (float*)alloc((size_t)CD * 4);
  float*  rmax    = (float*)alloc((size_t)CD * 4);

  float* out_nodes = (float*)d_out;
  float* prop = out_nodes + (size_t)NN * CD;

  // --- weight prep ---
  k_wt<<<CEILDIV(256 * 32, 256), 256, 0, stream>>>(W_ac, Wac_t, 13, 256, 32);
  k_wt<<<CEILDIV(256 * 256, 256), 256, 0, stream>>>(Wl, Wl_t, 256, 256, 256);
  k_wt<<<CEILDIV(256 * 256, 256), 256, 0, stream>>>(Wr, Wr_t, 256, 256, 256);
  k_wt<<<CEILDIV(512 * 256, 256), 256, 0, stream>>>(W_gin, Wgin_t, 256, 512, 256);
  k_wt<<<CEILDIV(512 * 512, 256), 256, 0, stream>>>(W_th, Wth_t, 512, 512, 512);
  k_wt<<<CEILDIV(512 * 512, 256), 256, 0, stream>>>(W_ph, Wph_t, 512, 512, 512);
  k_wt<<<CEILDIV(512 * 512, 256), 256, 0, stream>>>(Wq, Wq_t, 512, 512, 512);
  k_wt<<<CEILDIV(512 * 512, 256), 256, 0, stream>>>(Wk, Wk_t, 512, 512, 512);
  k_wt<<<CEILDIV(512 * 512, 256), 256, 0, stream>>>(Wv, Wv_t, 512, 512, 512);
  k_wt<<<CEILDIV(512 * 512, 256), 256, 0, stream>>>(Wskip, Wskip_t, 512, 512, 512);
  for (int i = 0; i < 3; ++i) {
    k_wt<<<CEILDIV(512 * 512, 256), 256, 0, stream>>>(Wv_c + (size_t)i * 512 * 512,
                                                      Wvc_t + (size_t)i * 512 * 512, 512, 512, 512);
    k_wt<<<CEILDIV(512 * 512, 256), 256, 0, stream>>>(Wo_c + (size_t)i * 512 * 512,
                                                      Woc_t + (size_t)i * 512 * 512, 512, 512, 512);
  }
  k_addvec<<<2, 256, 0, stream>>>(b_th, b_ph, b_comb, CD);

  // --- per-motif GraphConv + GATv2 (WMMA) ---
  k_motif<<<NN, 32, 0, stream>>>(atom_x, motif_ew, motif_src, motif_dst,
                                 Wac_t, b_ac, Wl_t, Wr_t, attn_a, b_gat, mf, mf_bf);

  // --- GIN ---
  k_fillf<<<CEILDIV(NN * MD, 256), 256, 0, stream>>>(agg, 0.f, NN * MD);
  k_gin_agg<<<NE, MD, 0, stream>>>(mf, chain_src, chain_dst, agg, NE);
  k_add_conv<<<CEILDIV(NN * MD, 256), 256, 0, stream>>>(mf, agg, s_bf, NN * MD);
  k_gemm<<<dim3(NN / 16, CD / 64), 32, 0, stream>>>(s_bf, Wgin_t, b_gin, hgin, hgin_bf, MD, CD, 0);

  // --- EdgeConv (factored): t = hgin@W_th, p = hgin@W_ph + (b_th + b_ph) ---
  k_gemm<<<dim3(NN / 16, CD / 64), 32, 0, stream>>>(hgin_bf, Wth_t, nullptr, t_th, nullptr, CD, CD, 0);
  k_gemm<<<dim3(NN / 16, CD / 64), 32, 0, stream>>>(hgin_bf, Wph_t, b_comb, p_ph, nullptr, CD, CD, 0);
  k_fillf<<<CEILDIV(NN * CD, 256), 256, 0, stream>>>(h_edge, -INFINITY, NN * CD);
  k_edge_max<<<NE, 256, 0, stream>>>(t_th, p_ph, chain_src, chain_dst, h_edge, NE);
  k_edge_fix<<<CEILDIV(NN * CD, 256), 256, 0, stream>>>(h_edge, NN * CD);

  // --- TransformerConv ---
  k_gemm<<<dim3(NN / 16, CD / 64), 32, 0, stream>>>(hgin_bf, Wq_t, bq, qb, nullptr, CD, CD, 0);
  k_gemm<<<dim3(NN / 16, CD / 64), 32, 0, stream>>>(hgin_bf, Wk_t, bk, kb, nullptr, CD, CD, 0);
  k_gemm<<<dim3(NN / 16, CD / 64), 32, 0, stream>>>(hgin_bf, Wv_t, bvv, vb, nullptr, CD, CD, 0);
  k_fillf<<<CEILDIV(NN * 4, 256), 256, 0, stream>>>(mseg, -INFINITY, NN * 4);
  k_fillf<<<CEILDIV(NN * 4, 256), 256, 0, stream>>>(zseg, 0.f, NN * 4);
  k_fillf<<<CEILDIV(NN * CD, 256), 256, 0, stream>>>(h_tr, 0.f, NN * CD);
  k_tr_score<<<CEILDIV(NE, 8), 256, 0, stream>>>(qb, kb, chain_eattr, We, be,
                                                 chain_src, chain_dst, sc, NE);
  k_tr_max<<<CEILDIV(NE * 4, 256), 256, 0, stream>>>(sc, chain_dst, mseg, NE);
  k_tr_exp<<<CEILDIV(NE * 4, 256), 256, 0, stream>>>(sc, chain_dst, mseg, zseg, NE);
  k_tr_agg<<<CEILDIV(NE, 8), 256, 0, stream>>>(vb, chain_eattr, We, be, chain_src,
                                               chain_dst, sc, zseg, h_tr, NE);
  // h_tr += hgin @ Wskip + bskip  (accumulating GEMM)
  k_gemm<<<dim3(NN / 16, CD / 64), 32, 0, stream>>>(hgin_bf, Wskip_t, bskip, h_tr, nullptr, CD, CD, 2);

  // --- virtual node + 3 residual cross-attn layers ---
  k_chinit<<<CEILDIV(NN * CD, 256), 256, 0, stream>>>(h_tr, vnode, ch, ch_bf, NN * CD);
  for (int i = 0; i < 3; ++i) {
    k_gemm<<<dim3(NN / 16, CD / 64), 32, 0, stream>>>(ch_bf, Wvc_t + (size_t)i * 512 * 512,
                                                      bv_c + (size_t)i * 512, tmp, tmp_bf, CD, CD, 0);
    k_gemm<<<dim3(NN / 16, CD / 64), 32, 0, stream>>>(tmp_bf, Woc_t + (size_t)i * 512 * 512,
                                                      bo_c + (size_t)i * 512, ch, ch_bf, CD, CD, 2);
  }

  // --- gated fusion -> out nodes ---
  k_gate<<<CEILDIV(NN, 8), 256, 0, stream>>>(hgin, h_edge, h_tr, ch, W_gate, b_gate, out_nodes, NN);

  // --- readout + head ---
  k_fillf<<<CEILDIV(CD, 256), 256, 0, stream>>>(rsum, 0.f, CD);
  k_fillf<<<CEILDIV(CD, 256), 256, 0, stream>>>(rmax, -INFINITY, CD);
  k_reduce<<<CEILDIV(NN * CD, 256), 256, 0, stream>>>(out_nodes, rsum, rmax, NN * CD);
  k_head<<<1, 1024, 0, stream>>>(rsum, rmax, W1, b1, bn_g, bn_b, W2, b2, prop, 1.f / (float)NN);
}
